// GNN_32427003085209
// MI455X (gfx1250) — compile-verified, matrix operand order checked by reading, not run on hardware
//
#include <hip/hip_runtime.h>
#include <hip/hip_bf16.h>

typedef __attribute__((ext_vector_type(16))) __bf16 v16bf;
typedef __attribute__((ext_vector_type(8)))  float  v8f;
typedef __attribute__((ext_vector_type(2)))  float  v2f;
typedef __attribute__((ext_vector_type(4)))  unsigned int v4u;
typedef __attribute__((ext_vector_type(8)))  int    v8i;
typedef __attribute__((ext_vector_type(4)))  int    v4i;

#define NN   100000   // nodes
#define NE   1600000  // edges
#define F    128      // feature dim (both GCN layers)
#define P    32       // SOPOOL dim
#define NG   100      // graphs
#define CDIV(a,b) (((a)+(b)-1)/(b))

// ---------------------------------------------------------------- degree
__global__ void k_deg_init(float* deg) {
    int i = blockIdx.x * 256 + threadIdx.x;
    if (i < NN) deg[i] = 1.0f;                       // self-loop weight
}
__global__ void k_deg_acc(const int* __restrict__ dst, const float* __restrict__ ew,
                          float* deg) {
    int e = blockIdx.x * 256 + threadIdx.x;
    if (e < NE) atomicAdd(&deg[dst[e]], ew[e]);
}
__global__ void k_rsqrt(float* dinv) {
    int i = blockIdx.x * 256 + threadIdx.x;
    if (i < NN) dinv[i] = rsqrtf(dinv[i]);           // deg >= 1, safe
}

// ------------------------------------------------- GEMM  H = X @ W  (N x 128 @ 128 x 128)
// bf16 WMMA, 8 waves/block, each wave: 16-row strip, 8 x (16x16) tiles, K in 4 steps of 32.
__device__ __forceinline__ int kmap(int e, int hi) {
    // 16-bit WMMA operand K index for vector element e (0..15), lane-half hi
    return e + ((e < 8) ? 0 : 8) + (hi ? 8 : 0);
}

__global__ void __launch_bounds__(256)
k_gemm128(const float* __restrict__ X, const float* __restrict__ W,
          float* __restrict__ H) {
    // LDS: W pre-swizzled into exact B-operand layout: [kb32][ntile][lane] -> 16 bf16
    __shared__ v16bf ldsB[4 * 8 * 32];               // 32 KB
    {
        __bf16* l = (__bf16*)ldsB;
        for (int i = threadIdx.x; i < 4 * 8 * 32 * 16; i += 256) {
            int e    = i & 15;
            int lane = (i >> 4) & 31;
            int nt   = (i >> 9) & 7;
            int kb32 = i >> 12;
            int k = kb32 * 32 + kmap(e, lane >> 4);
            int n = nt * 16 + (lane & 15);
            l[i] = (__bf16)W[k * F + n];
        }
    }
    __syncthreads();

    const int wave = threadIdx.x >> 5;
    const int lane = threadIdx.x & 31;
    const int hi   = lane >> 4;
    const int m    = lane & 15;
    const long row0 = (long)blockIdx.x * 128 + wave * 16;
    long r  = row0 + m;
    long rc = (r < NN) ? r : (NN - 1);               // clamp loads, guard stores
    const float* xr = X + rc * F;

    v8f c[8] = {};
#pragma unroll
    for (int kb = 0; kb < 4; ++kb) {
        // A operand: two contiguous 8-float runs per lane (documented 16-bit A layout)
        int base = kb * 32 + hi * 8;
        float4 f0 = *(const float4*)(xr + base);
        float4 f1 = *(const float4*)(xr + base + 4);
        float4 f2 = *(const float4*)(xr + base + 16);
        float4 f3 = *(const float4*)(xr + base + 20);
        v16bf a;
        a[0] = (__bf16)f0.x; a[1] = (__bf16)f0.y; a[2] = (__bf16)f0.z; a[3] = (__bf16)f0.w;
        a[4] = (__bf16)f1.x; a[5] = (__bf16)f1.y; a[6] = (__bf16)f1.z; a[7] = (__bf16)f1.w;
        a[8]  = (__bf16)f2.x; a[9]  = (__bf16)f2.y; a[10] = (__bf16)f2.z; a[11] = (__bf16)f2.w;
        a[12] = (__bf16)f3.x; a[13] = (__bf16)f3.y; a[14] = (__bf16)f3.z; a[15] = (__bf16)f3.w;
        // software-pipeline the B-operand LDS reads one tile ahead of the matrix pipe
        v16bf bcur = ldsB[(kb * 8) * 32 + lane];
#pragma unroll
        for (int nt = 0; nt < 8; ++nt) {
            v16bf bnext = ldsB[(kb * 8 + ((nt + 1) & 7)) * 32 + lane];
            c[nt] = __builtin_amdgcn_wmma_f32_16x16x32_bf16(
                        false, a, false, bcur, (short)0, c[nt], false, false);
            bcur = bnext;
        }
    }
#pragma unroll
    for (int nt = 0; nt < 8; ++nt) {
        int col = nt * 16 + m;
#pragma unroll
        for (int rr = 0; rr < 8; ++rr) {
            long orow = row0 + rr + 8 * hi;          // C layout: hi lanes hold M+8
            if (orow < NN) H[orow * F + col] = c[nt][rr];
        }
    }
}

// ------------------------------------------------- aggregation (self-loop + edges + bias/relu)
__global__ void k_self(const float* __restrict__ T, const float* __restrict__ dinv,
                       float* __restrict__ AGG) {
    long i = (long)blockIdx.x * 256 + threadIdx.x;
    if (i < (long)NN * F) {
        long n = i >> 7;
        float d = dinv[n];
        AGG[i] = T[i] * d * d;
    }
}
__global__ void k_edge(const int* __restrict__ src, const int* __restrict__ dst,
                       const float* __restrict__ ew, const float* __restrict__ dinv,
                       const float* __restrict__ T, float* __restrict__ AGG) {
    int w    = (blockIdx.x * 256 + threadIdx.x) >> 5;   // one wave per edge
    int lane = threadIdx.x & 31;
    if (w >= NE) return;
    int s = src[w], d = dst[w];
    float nrm = dinv[s] * ew[w] * dinv[d];
    float4 v = ((const float4*)(T + (long)s * F))[lane]; // lane covers 4 features
    float* ad = AGG + (long)d * F + lane * 4;
    atomicAdd(ad + 0, v.x * nrm);
    atomicAdd(ad + 1, v.y * nrm);
    atomicAdd(ad + 2, v.z * nrm);
    atomicAdd(ad + 3, v.w * nrm);
}
__global__ void k_bias_relu(const float* __restrict__ AGG, const float* __restrict__ b,
                            float* __restrict__ H) {
    long i = (long)blockIdx.x * 256 + threadIdx.x;
    if (i < (long)NN * F) {
        float v = AGG[i] + b[i & 127];
        H[i] = v > 0.f ? v : 0.f;
    }
}

// ------------------------------------------------- fused normalize + SOPOOL node MLP
__global__ void __launch_bounds__(256)
k_sopool(const float* __restrict__ H2,
         const float* __restrict__ sW1, const float* __restrict__ sb1,
         const float* __restrict__ sW2, const float* __restrict__ sb2,
         const float* __restrict__ sW3, const float* __restrict__ sb3,
         float* __restrict__ G3) {
    __shared__ float w1[F * P], w2[P * P], w3[P * P], bb1[P], bb2[P], bb3[P];
    for (int i = threadIdx.x; i < F * P; i += 256) w1[i] = sW1[i];
    for (int i = threadIdx.x; i < P * P; i += 256) { w2[i] = sW2[i]; w3[i] = sW3[i]; }
    if (threadIdx.x < P) {
        bb1[threadIdx.x] = sb1[threadIdx.x];
        bb2[threadIdx.x] = sb2[threadIdx.x];
        bb3[threadIdx.x] = sb3[threadIdx.x];
    }
    __syncthreads();
    int wave = threadIdx.x >> 5, lane = threadIdx.x & 31;
    long node = (long)blockIdx.x * 8 + wave;
    if (node >= NN) return;
    const float* h = H2 + node * F;
    // row L2 norm: per-lane partial over 4 feats, wave32 butterfly reduce
    float4 hv = ((const float4*)h)[lane];
    float ss = hv.x * hv.x + hv.y * hv.y + hv.z * hv.z + hv.w * hv.w;
#pragma unroll
    for (int off = 16; off; off >>= 1) ss += __shfl_xor(ss, off, 32);
    float scale = 1.0f / fmaxf(sqrtf(ss), 1e-12f);
    // layer 1: lane = output feature
    float acc = bb1[lane];
#pragma unroll 8
    for (int k = 0; k < F; ++k) acc += h[k] * scale * w1[k * P + lane];
    float g1 = fmaxf(acc, 0.f);
    float a2 = bb2[lane];
#pragma unroll
    for (int k = 0; k < P; ++k) a2 += __shfl(g1, k, 32) * w2[k * P + lane];
    float g2 = fmaxf(a2, 0.f);
    float a3 = bb3[lane];
#pragma unroll
    for (int k = 0; k < P; ++k) a3 += __shfl(g2, k, 32) * w3[k * P + lane];
    G3[node * P + lane] = fmaxf(a3, 0.f);
}

// ------------------------------------------------- TDM: async 32x32 f32 tile -> LDS
// 2D D#: data_size=4B, tensor_dim0=32 (feats), tensor_dim1=rows_remaining (OOB rows
// zero-fill the tail), tile 32x32, dim0 stride = 32. Groups 2/3 zero (2D tensor).
__device__ __forceinline__ void tdm_load_tile(const float* gbase, unsigned lds_off,
                                              int rows_remaining) {
    unsigned long long ga = (unsigned long long)(uintptr_t)gbase;
    v4u g0;
    g0[0] = 1u;                                          // count=1, user mode
    g0[1] = lds_off;                                     // LDS byte address
    g0[2] = (unsigned)(ga & 0xFFFFFFFFu);                // global_addr lo
    g0[3] = (unsigned)((ga >> 32) & 0x01FFFFFFu) | (2u << 30); // addr hi | type=2
    unsigned td1 = (unsigned)rows_remaining;
    v8i g1;
    g1[0] = (int)(2u << 16);                             // data_size = 4 bytes
    g1[1] = (int)(32u << 16);                            // tensor_dim0 = 32 (lo16)
    g1[2] = (int)((td1 & 0xFFFFu) << 16);                // dim0 hi16=0 | tensor_dim1 lo16
    g1[3] = (int)(((td1 >> 16) & 0xFFFFu) | (32u << 16));// tensor_dim1 hi16 | tile_dim0=32
    g1[4] = (int)32u;                                    // tile_dim1=32, tile_dim2=0
    g1[5] = (int)32u;                                    // tensor_dim0_stride = 32
    g1[6] = 0;
    g1[7] = 0;
    v4i z4 = {0, 0, 0, 0};
    v8i z8 = {0, 0, 0, 0, 0, 0, 0, 0};
    // 6-arg variant (clang-23 / therock headers): groups 0..3 + extra group + cpol
    __builtin_amdgcn_tensor_load_to_lds(g0, g1, z4, z4, z8, 0);
}

// ------------------------------------------------- per-graph HH = G^T G via f32 WMMA 16x16x4
// TDM double-buffered: wave 0 issues the next 32-node tile while all 4 waves run WMMAs.
__global__ void __launch_bounds__(128)
k_hh(const float* __restrict__ G3, const int* __restrict__ ptr,
     float* __restrict__ HH) {
    __shared__ __align__(16) float tile[2][32 * 32];     // double-buffered 32 nodes x 32 feats
    int g  = blockIdx.x;
    int n0 = ptr[g], n1 = ptr[g + 1];
    int nn = n1 - n0;
    int nchunk = (nn + 31) >> 5;
    int wave = threadIdx.x >> 5, lane = threadIdx.x & 31;
    int hi = lane >> 4, l15 = lane & 15;
    int mi = wave >> 1, ni = wave & 1;                   // 4 waves -> 4 16x16 tiles of 32x32
    int mrow = mi * 16 + l15, ncol = ni * 16 + l15;
    unsigned lds0 = (unsigned)(uintptr_t)&tile[0][0];    // generic addr[31:0] == LDS offset
    unsigned lds1 = (unsigned)(uintptr_t)&tile[1][0];

    if (wave == 0 && nchunk > 0) {
        tdm_load_tile(G3 + (long)n0 * P, lds0, nn);
        __builtin_amdgcn_s_wait_tensorcnt(0);
    }
    __syncthreads();

    v8f c = {};
    for (int ci = 0; ci < nchunk; ++ci) {
        int buf = ci & 1;
        if (wave == 0 && ci + 1 < nchunk)                // prefetch next tile into other buffer
            tdm_load_tile(G3 + (long)(n0 + (ci + 1) * 32) * P,
                          buf ? lds0 : lds1, nn - (ci + 1) * 32);
        const float* t = &tile[buf][0];
#pragma unroll
        for (int s = 0; s < 8; ++s) {
            int kb = s * 4;
            v2f a, b;                                    // f32 A 16x4: lane<16 K=0/1, hi K=2/3
            a.x = t[(kb + (hi ? 2 : 0)) * 32 + mrow];
            a.y = t[(kb + (hi ? 3 : 1)) * 32 + mrow];
            b.x = t[(kb + (hi ? 2 : 0)) * 32 + ncol];
            b.y = t[(kb + (hi ? 3 : 1)) * 32 + ncol];
            c = __builtin_amdgcn_wmma_f32_16x16x4_f32(
                    false, a, false, b, (short)0, c, false, false);
        }
        if (wave == 0 && ci + 1 < nchunk)
            __builtin_amdgcn_s_wait_tensorcnt(0);
        __syncthreads();
    }
    float* out = HH + (long)g * (P * P);
#pragma unroll
    for (int rr = 0; rr < 8; ++rr) {
        int row = mi * 16 + rr + 8 * hi;
        out[row * P + ncol] = c[rr];
    }
}

// ------------------------------------------------- graph MLP on HH (100 x 1024 -> 2)
__global__ void k_mlp(const float* __restrict__ HH,
                      const float* __restrict__ mW1, const float* __restrict__ mb1,
                      const float* __restrict__ mW2, const float* __restrict__ mb2,
                      const float* __restrict__ mW3, const float* __restrict__ mb3,
                      float* __restrict__ O) {
    int wave = threadIdx.x >> 5, lane = threadIdx.x & 31;
    int g = blockIdx.x * 8 + wave;
    if (g >= NG) return;
    const float* hh = HH + (long)g * (P * P);
    float acc = mb1[lane];
#pragma unroll 8
    for (int k = 0; k < P * P; ++k) acc += hh[k] * mW1[k * 32 + lane];
    float h1 = fmaxf(acc, 0.f);
    float a2 = mb2[lane];
#pragma unroll
    for (int k = 0; k < 32; ++k) a2 += __shfl(h1, k, 32) * mW2[k * 32 + lane];
    float h2 = fmaxf(a2, 0.f);
    float o0 = mb3[0], o1 = mb3[1];
#pragma unroll
    for (int k = 0; k < 32; ++k) {
        float v = __shfl(h2, k, 32);
        o0 += v * mW3[k * 2 + 0];
        o1 += v * mW3[k * 2 + 1];
    }
    if (lane == 0) { O[g * 2 + 0] = fmaxf(o0, 0.f); O[g * 2 + 1] = fmaxf(o1, 0.f); }
}

// ----------------------------------------------------------------- launch
extern "C" void kernel_launch(void* const* d_in, const int* in_sizes, int n_in,
                              void* d_out, int out_size, void* d_ws, size_t ws_size,
                              hipStream_t stream) {
    const float* x   = (const float*)d_in[0];
    const int*   ei  = (const int*)d_in[1];
    const float* ew  = (const float*)d_in[2];
    const int*   ptr = (const int*)d_in[3];
    const float* W1  = (const float*)d_in[4];  const float* b1  = (const float*)d_in[5];
    const float* W2  = (const float*)d_in[6];  const float* b2  = (const float*)d_in[7];
    const float* sW1 = (const float*)d_in[8];  const float* sb1 = (const float*)d_in[9];
    const float* sW2 = (const float*)d_in[10]; const float* sb2 = (const float*)d_in[11];
    const float* sW3 = (const float*)d_in[12]; const float* sb3 = (const float*)d_in[13];
    const float* mW1 = (const float*)d_in[14]; const float* mb1 = (const float*)d_in[15];
    const float* mW2 = (const float*)d_in[16]; const float* mb2 = (const float*)d_in[17];
    const float* mW3 = (const float*)d_in[18]; const float* mb3 = (const float*)d_in[19];
    const int* src = ei;
    const int* dst = ei + NE;

    // workspace layout (256B aligned slabs)
    char* ws = (char*)d_ws;
    size_t off = 0;
    auto alloc = [&](size_t bytes) { char* p = ws + off; off = (off + bytes + 255) & ~(size_t)255; return p; };
    float* dinv = (float*)alloc(sizeof(float) * NN);
    float* bufA = (float*)alloc(sizeof(float) * (size_t)NN * F);   // GEMM outputs t1/t2
    float* bufB = (float*)alloc(sizeof(float) * (size_t)NN * F);   // agg / h1 / h2
    float* g3   = (float*)alloc(sizeof(float) * (size_t)NN * P);   // SOPOOL features

    float* HHout = (float*)d_out;                   // [NG, 1024]
    float* Oout  = (float*)d_out + NG * P * P;      // [NG, 2]

    // degrees -> dinv
    k_deg_init<<<CDIV(NN, 256), 256, 0, stream>>>(dinv);
    k_deg_acc <<<CDIV(NE, 256), 256, 0, stream>>>(dst, ew, dinv);
    k_rsqrt   <<<CDIV(NN, 256), 256, 0, stream>>>(dinv);

    // layer 1: t1 = x@W1 ; agg = scatter + self ; h1 = relu(agg + b1)
    k_gemm128 <<<CDIV(NN, 128), 256, 0, stream>>>(x, W1, bufA);
    k_self    <<<CDIV(NN * F, 256), 256, 0, stream>>>(bufA, dinv, bufB);
    k_edge    <<<NE / 8, 256, 0, stream>>>(src, dst, ew, dinv, bufA, bufB);
    k_bias_relu<<<CDIV(NN * F, 256), 256, 0, stream>>>(bufB, b1, bufB);

    // layer 2
    k_gemm128 <<<CDIV(NN, 128), 256, 0, stream>>>(bufB, W2, bufA);
    k_self    <<<CDIV(NN * F, 256), 256, 0, stream>>>(bufA, dinv, bufB);
    k_edge    <<<NE / 8, 256, 0, stream>>>(src, dst, ew, dinv, bufA, bufB);
    k_bias_relu<<<CDIV(NN * F, 256), 256, 0, stream>>>(bufB, b2, bufB);

    // normalize + node MLP (fused), then per-graph G^T G (TDM + WMMA), then graph MLP
    k_sopool  <<<CDIV(NN, 8), 256, 0, stream>>>(bufB, sW1, sb1, sW2, sb2, sW3, sb3, g3);
    k_hh      <<<NG, 128, 0, stream>>>(g3, ptr, HHout);
    k_mlp     <<<CDIV(NG, 8), 256, 0, stream>>>(HHout, mW1, mb1, mW2, mb2, mW3, mb3, Oout);
}
